// LinDense_48644799594600
// MI455X (gfx1250) — matrix operand order repaired and checked by reading, not compiled, not run on hardware
//
#include <hip/hip_runtime.h>
#include <hip/hip_bf16.h>

typedef __attribute__((ext_vector_type(16))) _Float16 v16h;
typedef __attribute__((ext_vector_type(8)))  float    v8f;
typedef __attribute__((ext_vector_type(4)))  float    v4f;

// Types for the async-copy builtin: pointee is a GCC vector of 4 ints,
// global side in AS1 ("__device__"), LDS side in AS3.
typedef int v4i_vec __attribute__((vector_size(16)));
typedef __attribute__((address_space(1))) v4i_vec* gptr_v4i;
typedef __attribute__((address_space(3))) v4i_vec* lptr_v4i;

#define WAVES_PER_WG 4
#define NN 64

#if defined(__has_builtin)
#if __has_builtin(__builtin_amdgcn_global_load_async_to_lds_b128)
#define HAVE_ASYNC_LDS 1
#endif
#endif

__device__ __forceinline__ void wait_asynccnt0() {
    asm volatile("s_wait_asynccnt 0" ::: "memory");
}

// ---------------------------------------------------------------------------
// Setup kernel (1 block, 256 threads, runs once per launch): pre-convert the
// shared 64x64 weights to f16 and store them in the exact per-lane VGPR image
// of the WMMA operands, so every compute wave loads each 8-VGPR tile with one
// 32B vector load instead of 16 scalar loads + 8 cvt_pk.
//   ws[0      .. 4095]: W0^T tiles, A-layout, tile = i*2+kt, [tile][lane][e]
//   ws[4096   .. 8191]: W1   tiles, B-layout, tile = kt*4+j, [tile][lane][e]
// ---------------------------------------------------------------------------
__global__ void pack_weights_kernel(const float* __restrict__ W0,
                                    const float* __restrict__ W1,
                                    _Float16* __restrict__ ws)
{
    const int t    = threadIdx.x;      // 0..255
    const int tile = t >> 5;           // 0..7
    const int lane = t & 31;
    const int half = lane >> 4;
    const int l16  = lane & 15;

    // W0^T -> A-layout (16x32 f16): row m = l16; elem e -> K = (e<8?e:e+8)+8*half
    {
        const int i  = tile >> 1;
        const int kt = tile & 1;
        const int m  = i * 16 + l16;                   // a' index
        _Float16* dst = ws + ((size_t)tile * 32 + lane) * 16;
        #pragma unroll
        for (int e = 0; e < 16; ++e) {
            const int K = kt * 32 + (e < 8 ? e : e + 8) + half * 8;   // a index
            dst[e] = (_Float16)W0[K * NN + m];         // W0T[m,K] = W0[K,m]
        }
    }
    // W1 -> B-layout (32x16 f16): col n = l16; elem e -> K = (e<8?e:e+8)+8*half
    {
        const int kt = tile >> 2;
        const int j  = tile & 3;
        const int n  = j * 16 + l16;                   // bdim' index
        _Float16* dst = ws + 4096 + ((size_t)tile * 32 + lane) * 16;
        #pragma unroll
        for (int e = 0; e < 16; ++e) {
            const int K = kt * 32 + (e < 8 ? e : e + 8) + half * 8;     // bdim index
            dst[e] = (_Float16)W1[K * NN + n];
        }
    }
}

// ---------------------------------------------------------------------------
// Main kernel: OUT = H1 + H1*W1 + b1,  H1 = M + W0^T*M + b0,  M = x[b] 64x64.
// One batch per wave; per-wave private 16KB LDS tile; no cross-wave sync.
// ---------------------------------------------------------------------------
__launch_bounds__(128, 2)
__global__ void lindense_wmma_kernel(const float* __restrict__ x,
                                     const _Float16* __restrict__ wpack,
                                     const float* __restrict__ b0,
                                     const float* __restrict__ b1,
                                     float* __restrict__ out,
                                     int n_batch)
{
    __shared__ float lds[WAVES_PER_WG][NN * NN];   // 64 KB
    const int lane  = threadIdx.x & 31;
    const int wave  = threadIdx.x >> 5;
    const int batch = blockIdx.x * WAVES_PER_WG + wave;
    const int half  = lane >> 4;    // 0: lanes 0-15, 1: lanes 16-31
    const int l16   = lane & 15;
    float* buf = lds[wave];
    const bool active = (batch < n_batch);

    // ---- kick off async global -> LDS copy of M = x[batch] (ASYNCcnt-tracked) ----
#ifdef HAVE_ASYNC_LDS
    if (active) {
        gptr_v4i gsrc = (gptr_v4i)(x + (size_t)batch * (NN * NN));
        lptr_v4i ldst = (lptr_v4i)buf;
        #pragma unroll
        for (int t = 0; t < 32; ++t) {
            __builtin_amdgcn_global_load_async_to_lds_b128(
                gsrc + (t * 32 + lane),
                ldst + (t * 32 + lane),
                0, 0);
        }
    }
#else
    if (active) {
        const v4f* src = (const v4f*)(x + (size_t)batch * (NN * NN));
        v4f* dst = (v4f*)buf;
        #pragma unroll
        for (int t = 0; t < 32; ++t)
            dst[t * 32 + lane] = src[t * 32 + lane];
    }
#endif

    // ---- W0^T A-layout tiles: one 32B vector load per tile (overlaps async copy) ----
    const v16h* wsA = (const v16h*)wpack;
    v16h WA[4][2];
    #pragma unroll
    for (int i = 0; i < 4; ++i)
        #pragma unroll
        for (int kt = 0; kt < 2; ++kt)
            WA[i][kt] = wsA[(i * 2 + kt) * 32 + lane];

#ifdef HAVE_ASYNC_LDS
    wait_asynccnt0();
#endif

    // ============ stage A: H1 = M + b0 + W0^T*M, two column-halves ============
    #pragma unroll
    for (int jh = 0; jh < 2; ++jh) {
        v8f C[4][2];
        // init: C = M + b0   (f32 C layout: col = l16, VGPR r -> row r + 8*half)
        #pragma unroll
        for (int i = 0; i < 4; ++i) {
            #pragma unroll
            for (int jj = 0; jj < 2; ++jj) {
                v8f c;
                const int col = (jh * 2 + jj) * 16 + l16;
                #pragma unroll
                for (int r = 0; r < 8; ++r) {
                    const int row = i * 16 + r + half * 8;   // a'
                    c[r] = buf[row * NN + col] + b0[row];
                }
                C[i][jj] = c;
            }
        }
        // accumulate W0^T * M over K
        #pragma unroll
        for (int kt = 0; kt < 2; ++kt) {
            #pragma unroll
            for (int jj = 0; jj < 2; ++jj) {
                v16h mb;   // M tile in B layout (K = a rows, N = bdim cols), f16
                const int n = (jh * 2 + jj) * 16 + l16;
                #pragma unroll
                for (int e = 0; e < 16; ++e) {
                    const int K = kt * 32 + (e < 8 ? e : e + 8) + half * 8;
                    mb[e] = (_Float16)buf[K * NN + n];
                }
                #pragma unroll
                for (int i = 0; i < 4; ++i)
                    C[i][jj] = __builtin_amdgcn_wmma_f32_16x16x32_f16(
                        false, WA[i][kt], false, mb, (short)0, C[i][jj], false, false);
            }
        }
        // write H1 columns of this half back to LDS (same-wave DS ops stay in order)
        #pragma unroll
        for (int i = 0; i < 4; ++i) {
            #pragma unroll
            for (int jj = 0; jj < 2; ++jj) {
                const int col = (jh * 2 + jj) * 16 + l16;
                #pragma unroll
                for (int r = 0; r < 8; ++r) {
                    const int row = i * 16 + r + half * 8;
                    buf[row * NN + col] = C[i][jj][r];
                }
            }
        }
    }

    // ---- W1 B-layout tiles (WA now dead): one 32B vector load per tile ----
    const v16h* wsB = (const v16h*)(wpack + 4096);
    v16h WB[2][4];
    #pragma unroll
    for (int kt = 0; kt < 2; ++kt)
        #pragma unroll
        for (int j = 0; j < 4; ++j)
            WB[kt][j] = wsB[(kt * 4 + j) * 32 + lane];

    // ============ stage B: OUT = H1 + b1 + H1*W1, two row-halves ============
    #pragma unroll
    for (int ih = 0; ih < 2; ++ih) {
        v8f C[2][4];
        // init: C = H1 + b1 (re-read from LDS)
        #pragma unroll
        for (int ii = 0; ii < 2; ++ii) {
            #pragma unroll
            for (int j = 0; j < 4; ++j) {
                v8f c;
                const int col = j * 16 + l16;
                const float bb = b1[col];
                #pragma unroll
                for (int r = 0; r < 8; ++r) {
                    const int row = (ih * 2 + ii) * 16 + r + half * 8;   // a
                    c[r] = buf[row * NN + col] + bb;
                }
                C[ii][j] = c;
            }
        }
        // accumulate H1 * W1 over K (= bdim)
        #pragma unroll
        for (int kt = 0; kt < 2; ++kt) {
            #pragma unroll
            for (int ii = 0; ii < 2; ++ii) {
                v16h ha;   // H1 tile in A layout (rows = a, K = bdim), f16
                const int m = (ih * 2 + ii) * 16 + l16;
                #pragma unroll
                for (int e = 0; e < 16; ++e) {
                    const int K = kt * 32 + (e < 8 ? e : e + 8) + half * 8;
                    ha[e] = (_Float16)buf[m * NN + K];
                }
                #pragma unroll
                for (int j = 0; j < 4; ++j)
                    C[ii][j] = __builtin_amdgcn_wmma_f32_16x16x32_f16(
                        false, ha, false, WB[kt][j], (short)0, C[ii][j], false, false);
            }
        }
        // store this row-half of OUT
        if (active) {
            float* o = out + (size_t)batch * (NN * NN);
            #pragma unroll
            for (int ii = 0; ii < 2; ++ii) {
                #pragma unroll
                for (int j = 0; j < 4; ++j) {
                    const int col = j * 16 + l16;
                    #pragma unroll
                    for (int r = 0; r < 8; ++r) {
                        const int row = (ih * 2 + ii) * 16 + r + half * 8;
                        o[row * NN + col] = C[ii][j][r];
                    }
                }
            }
        }
    }
}

extern "C" void kernel_launch(void* const* d_in, const int* in_sizes, int n_in,
                              void* d_out, int out_size, void* d_ws, size_t ws_size,
                              hipStream_t stream) {
    const float* x  = (const float*)d_in[0];
    const float* W0 = (const float*)d_in[1];
    const float* W1 = (const float*)d_in[2];
    const float* b0 = (const float*)d_in[3];
    const float* b1 = (const float*)d_in[4];
    float* out = (float*)d_out;
    _Float16* wpack = (_Float16*)d_ws;     // 16 KB used

    pack_weights_kernel<<<1, 256, 0, stream>>>(W0, W1, wpack);

    const int n_batch = in_sizes[0] / (NN * NN);   // B = 16384
    const int blocks  = (n_batch + WAVES_PER_WG - 1) / WAVES_PER_WG;
    lindense_wmma_kernel<<<blocks, WAVES_PER_WG * 32, 0, stream>>>(
        x, wpack, b0, b1, out, n_batch);
}
